// ConditionalARNet_24421184045644
// MI455X (gfx1250) — matrix operand order repaired and checked by reading, not compiled, not run on hardware
//
#include <hip/hip_runtime.h>

typedef __attribute__((ext_vector_type(16))) _Float16 v16h;
typedef __attribute__((ext_vector_type(8)))  _Float16 v8h;
typedef __attribute__((ext_vector_type(8)))  float    v8f;
typedef __attribute__((ext_vector_type(4)))  float    v4f;

#define WAVES_PER_BLOCK 8
#define BLOCK (WAVES_PER_BLOCK * 32)

// Compiler + wave-level LDS ordering fence (DS ops execute in-order per wave).
#define WAVE_LDS_FENCE() __asm__ volatile("s_wait_dscnt 0" ::: "memory")

// 16 halfs from 16B-aligned LDS as two b128 loads (avoids 32B-align requirement).
__device__ __forceinline__ v16h ld16h(const _Float16* p) {
    v8h a = *(const v8h*)p;
    v8h b = *(const v8h*)(p + 16);
    return __builtin_shufflevector(a, b, 0, 1, 2, 3, 4, 5, 6, 7,
                                   8, 9, 10, 11, 12, 13, 14, 15);
}

// A[n][m] of the binarized adjacency (diag forced to 1), from 45 triu ints.
__device__ __forceinline__ float adj_entry(const int* __restrict__ a, int n, int m) {
    if (n == m) return 1.0f;
    int i = n < m ? n : m;
    int j = n < m ? m : n;
    int idx = i * 9 - ((i * (i - 1)) >> 1) + (j - i);   // triu_indices(9) incl. diag
    return a[idx] > 0 ? 1.0f : 0.0f;
}

__global__ __launch_bounds__(BLOCK) void fused_condar_kernel(
    const float* __restrict__ x,     const int* __restrict__ adj,
    const float* __restrict__ ctx,   const float* __restrict__ gcn_W,
    const float* __restrict__ gcn_b, const float* __restrict__ ctx_W,
    const float* __restrict__ ctx_b, const float* __restrict__ W1,
    const float* __restrict__ b1,    const float* __restrict__ W2,
    const float* __restrict__ b2,    const float* __restrict__ W3,
    const float* __restrict__ b3,    float* __restrict__ out, int Bn)
{
    // ---- block-shared weights ----
    __shared__ __attribute__((aligned(16))) float sCtxWT[63 * 48]; // [p][q], q padded 45->48 w/ 0
    __shared__ float sCtxB[63];
    __shared__ float sGcnW[49];
    __shared__ float sGcnB[7];
    __shared__ float sW1[128];            // (64,2) row-major
    __shared__ float sB1[64];
    __shared__ __attribute__((aligned(16))) _Float16 sW2h[64 * 72]; // W2 (n,k), row stride 72
    __shared__ __attribute__((aligned(16))) _Float16 sW3h[16 * 72]; // W3^T padded to 16 cols
    // ---- per-wave scratch ----
    __shared__ __attribute__((aligned(16))) float sCx[WAVES_PER_BLOCK][48]; // ctx padded w/ 0
    __shared__ float sX [WAVES_PER_BLOCK][64];
    __shared__ int   sAd[WAVES_PER_BLOCK][45];
    __shared__ float sXW[WAVES_PER_BLOCK][64];
    __shared__ float sD [WAVES_PER_BLOCK][9];
    __shared__ float sAn[WAVES_PER_BLOCK][81];
    __shared__ float sZ [WAVES_PER_BLOCK][64];
    __shared__ float sC [WAVES_PER_BLOCK][64];
    __shared__ __attribute__((aligned(16))) _Float16 sY[WAVES_PER_BLOCK][16 * 72]; // relu(h2) tile

    const int tid  = threadIdx.x;
    const int lane = tid & 31;
    const int wave = tid >> 5;
    const int half = lane >> 4;     // 0: lanes 0-15, 1: lanes 16-31
    const int l16  = lane & 15;

    for (int i = tid; i < 63 * 48; i += BLOCK) {
        int p = i / 48, q = i % 48;
        sCtxWT[i] = (q < 45) ? ctx_W[q * 63 + p] : 0.0f;
    }
    for (int i = tid; i < 63;  i += BLOCK) sCtxB[i] = ctx_b[i];
    for (int i = tid; i < 49;  i += BLOCK) sGcnW[i] = gcn_W[i];
    for (int i = tid; i < 7;   i += BLOCK) sGcnB[i] = gcn_b[i];
    for (int i = tid; i < 128; i += BLOCK) sW1[i]  = W1[i];
    for (int i = tid; i < 64;  i += BLOCK) sB1[i]  = b1[i];
    for (int i = tid; i < 4096; i += BLOCK) {
        int n = i >> 6, k = i & 63;
        sW2h[n * 72 + k] = (_Float16)W2[i];
    }
    for (int i = tid; i < 1024; i += BLOCK) {
        int n = i >> 6, k = i & 63;
        sW3h[n * 72 + k] = (n < 2) ? (_Float16)W3[n * 64 + k] : (_Float16)0.0f;
    }
    __syncthreads();

    // ---- per-lane layer-1 weights prepacked in f16 A-fragment element order ----
    // ISA 16-bit A 16x32: lane<16 -> K(e) = e + (e<8?0:8); lanes 16-31 add 8.
    v16h w1aH[2], w1bH[2], b1H[2];
#pragma unroll
    for (int ks = 0; ks < 2; ++ks)
#pragma unroll
        for (int e = 0; e < 16; ++e) {
            int K = ks * 32 + e + ((e & 8) ? 8 : 0) + (half ? 8 : 0);
            w1aH[ks][e] = (_Float16)sW1[2 * K];
            w1bH[ks][e] = (_Float16)sW1[2 * K + 1];
            b1H[ks][e]  = (_Float16)sB1[K];
        }
    const v16h vzeroh = {};

    // ---- hoist kernel-invariant B fragments (W2^T, W3^T) into registers ----
    // B f16 32x16 layout: lanes<16 hold K=0..7,16..23, lanes>=16 K=8..15,24..31, col n = l16.
    v16h w2f[2][4];   // [ks][nt]
#pragma unroll
    for (int ks = 0; ks < 2; ++ks) {
        const int krow = ks * 32 + (half ? 8 : 0);
#pragma unroll
        for (int nt = 0; nt < 4; ++nt)
            w2f[ks][nt] = ld16h(&sW2h[(nt * 16 + l16) * 72 + krow]);
    }
    v16h w3f[2];
#pragma unroll
    for (int ks = 0; ks < 2; ++ks)
        w3f[ks] = ld16h(&sW3h[l16 * 72 + ks * 32 + (half ? 8 : 0)]);

    // ---- per-lane layer-3 constants ----
    float b2v[4];
#pragma unroll
    for (int nt = 0; nt < 4; ++nt) b2v[nt] = b2[nt * 16 + l16];
    const float b3v = b3[l16 & 1];

    const int gwave = blockIdx.x * WAVES_PER_BLOCK + wave;
    const int nwave = gridDim.x * WAVES_PER_BLOCK;

    for (int el = gwave; el < Bn; el += nwave) {
        const float* xe  = x   + (size_t)el * 63;
        const float* cte = ctx + (size_t)el * 45;
        const int*   ae  = adj + (size_t)el * 45;

        // ---- stage per-element inputs ----
        sX[wave][lane] = xe[lane];
        if (lane + 32 < 63) sX[wave][lane + 32] = xe[lane + 32];
        if (lane == 31) { sX[wave][63] = 0.0f; sZ[wave][63] = 0.0f; sC[wave][63] = 0.0f; }
        sCx[wave][lane] = (lane < 45) ? cte[lane] : 0.0f;
        if (lane + 32 < 48) sCx[wave][lane + 32] = (lane + 32 < 45) ? cte[lane + 32] : 0.0f;
        if (lane < 45) sAd[wave][lane] = ae[lane];
        if (lane + 32 < 45) sAd[wave][lane + 32] = ae[lane + 32];
        WAVE_LDS_FENCE();

        // ---- xw = x @ gcn_W ----
        for (int p = lane; p < 63; p += 32) {
            int n = p / 7, f = p % 7;
            float s = 0.0f;
#pragma unroll
            for (int k = 0; k < 7; ++k) s += sX[wave][n * 7 + k] * sGcnW[k * 7 + f];
            sXW[wave][p] = s;
        }
        // ---- degree scaling d = rsqrt(max(rowsum,1)) ----
        if (lane < 9) {
            float s = 0.0f;
#pragma unroll
            for (int m = 0; m < 9; ++m) s += adj_entry(sAd[wave], lane, m);
            sD[wave][lane] = rsqrtf(fmaxf(s, 1.0f));
        }
        WAVE_LDS_FENCE();
        // ---- An = d_n * A * d_m ----
        for (int i = lane; i < 81; i += 32) {
            int n = i / 9, m = i % 9;
            sAn[wave][i] = sD[wave][n] * sD[wave][m] * adj_entry(sAd[wave], n, m);
        }
        WAVE_LDS_FENCE();
        // ---- z = An @ xw + gcn_b ; c = relu(ctx @ ctx_W + ctx_b) (vectorized) ----
        for (int p = lane; p < 63; p += 32) {
            int n = p / 7, f = p % 7;
            float s = sGcnB[f];
#pragma unroll
            for (int m = 0; m < 9; ++m) s += sAn[wave][n * 9 + m] * sXW[wave][m * 7 + f];
            sZ[wave][p] = s;

            float cc = sCtxB[p];
            const float* wr = &sCtxWT[p * 48];
#pragma unroll
            for (int chk = 0; chk < 48; chk += 4) {
                v4f a = *(const v4f*)&sCx[wave][chk];
                v4f w = *(const v4f*)&wr[chk];
                cc += a[0] * w[0] + a[1] * w[1] + a[2] * w[2] + a[3] * w[3];
            }
            sC[wave][p] = fmaxf(cc, 0.0f);
        }
        WAVE_LDS_FENCE();

        // ---- layers 1..3 : 4 row-tiles of 16 positions each (p=63 is padding) ----
        float* outB = out + (size_t)el * 126;
#pragma unroll 1
        for (int t = 0; t < 4; ++t) {
            const int p = t * 16 + l16;                 // this lane's A-matrix row
            const _Float16 zh = (_Float16)sZ[wave][p];
            const _Float16 ch = (_Float16)sC[wave][p];

            v8f acc[4];
#pragma unroll
            for (int nt = 0; nt < 4; ++nt) acc[nt] = v8f{};

#pragma unroll
            for (int ks = 0; ks < 2; ++ks) {
                // layer-1 output generated directly in f16 A-fragment layout (packed f16 fma)
                v16h af = w1aH[ks] * zh + w1bH[ks] * ch + b1H[ks];
                af = __builtin_elementwise_max(af, vzeroh);

#pragma unroll
                for (int nt = 0; nt < 4; ++nt) {
                    acc[nt] = __builtin_amdgcn_wmma_f32_16x16x32_f16(
                        false, af, false, w2f[ks][nt], (short)0, acc[nt], false, false);
                }
            }

            // ---- relu(h2 + b2) -> f16 LDS tile (row-major, stride 72) ----
#pragma unroll
            for (int nt = 0; nt < 4; ++nt)
#pragma unroll
                for (int j = 0; j < 8; ++j) {
                    float y = fmaxf(acc[nt][j] + b2v[nt], 0.0f);
                    sY[wave][(j + (half ? 8 : 0)) * 72 + nt * 16 + l16] = (_Float16)y;
                }
            WAVE_LDS_FENCE();

            // ---- layer 3 as WMMA: D = relu(h2) x W3^T(padded 64x16) ----
            v8f acc3 = v8f{};
            const _Float16* yr = &sY[wave][l16 * 72];
#pragma unroll
            for (int ks = 0; ks < 2; ++ks) {
                v8h c0 = *(const v8h*)&yr[ks * 32 + (half ? 8 : 0)];
                v8h c1 = *(const v8h*)&yr[ks * 32 + 16 + (half ? 8 : 0)];
                v16h a3 = __builtin_shufflevector(c0, c1, 0, 1, 2, 3, 4, 5, 6, 7,
                                                  8, 9, 10, 11, 12, 13, 14, 15);
                acc3 = __builtin_amdgcn_wmma_f32_16x16x32_f16(
                    false, a3, false, w3f[ks], (short)0, acc3, false, false);
            }

            // D layout: VGPR j -> row (j + 8*half), col = l16; only cols 0,1 are real.
#pragma unroll
            for (int j = 0; j < 8; ++j) {
                const int prow = t * 16 + j + (half ? 8 : 0);
                if (l16 < 2 && prow < 63) outB[l16 * 63 + prow] = acc3[j] + b3v;
            }
            WAVE_LDS_FENCE();
        }
    }
}

extern "C" void kernel_launch(void* const* d_in, const int* in_sizes, int n_in,
                              void* d_out, int out_size, void* d_ws, size_t ws_size,
                              hipStream_t stream) {
    const float* x     = (const float*)d_in[0];
    const int*   adj   = (const int*)  d_in[1];
    const float* ctx   = (const float*)d_in[2];
    const float* gcn_W = (const float*)d_in[3];
    const float* gcn_b = (const float*)d_in[4];
    const float* ctx_W = (const float*)d_in[5];
    const float* ctx_b = (const float*)d_in[6];
    const float* W1    = (const float*)d_in[7];
    const float* b1    = (const float*)d_in[8];
    const float* W2    = (const float*)d_in[9];
    const float* b2    = (const float*)d_in[10];
    const float* W3    = (const float*)d_in[11];
    const float* b3    = (const float*)d_in[12];
    float* out = (float*)d_out;

    const int Bn = in_sizes[0] / 63;               // x is (B,1,9,7)
    int grid = (Bn + WAVES_PER_BLOCK - 1) / WAVES_PER_BLOCK;
    if (grid > 1024) grid = 1024;                  // 8192 waves; grid-stride over elements
    if (grid < 1) grid = 1;

    fused_condar_kernel<<<grid, BLOCK, 0, stream>>>(
        x, adj, ctx, gcn_W, gcn_b, ctx_W, ctx_b, W1, b1, W2, b2, W3, b3, out, Bn);
}
